// graph_constructor_7610682048898
// MI455X (gfx1250) — matrix operand order repaired and checked by reading, not compiled, not run on hardware
//
#include <hip/hip_runtime.h>
#include <hip/hip_bf16.h>
#include <math.h>

typedef __attribute__((ext_vector_type(16))) _Float16 v16h;
typedef __attribute__((ext_vector_type(8)))  _Float16 v8h;
typedef __attribute__((ext_vector_type(8)))  float    v8f;

static constexpr int NN  = 8192;   // nodes
static constexpr int ND  = 64;     // dim
static constexpr int KSEL = 32;    // top-k

__device__ __forceinline__ float gelu_exact(float x) {
    // matches jax.nn.gelu(approximate=False): 0.5*x*(1+erf(x/sqrt(2)))
    return 0.5f * x * (1.0f + erff(x * 0.70710678118654752f));
}

// ---------------------------------------------------------------------------
// Kernel 1: v1 = gelu(emb @ W1^T + b1), v2 = gelu(emb @ W2^T + b2), stored f16.
// 4 rows per 256-thread block; W1/W2/b cached in LDS.
// ---------------------------------------------------------------------------
__global__ void mlp_kernel(const float* __restrict__ emb,
                           const float* __restrict__ W1, const float* __restrict__ b1,
                           const float* __restrict__ W2, const float* __restrict__ b2,
                           _Float16* __restrict__ v1h, _Float16* __restrict__ v2h) {
    __shared__ float W1s[ND * ND];
    __shared__ float W2s[ND * ND];
    __shared__ float b1s[ND], b2s[ND];
    __shared__ float es[4][ND];

    const int tid = threadIdx.x;
    for (int i = tid; i < ND * ND; i += 256) { W1s[i] = W1[i]; W2s[i] = W2[i]; }
    if (tid < ND) { b1s[tid] = b1[tid]; b2s[tid] = b2[tid]; }
    const int row0 = blockIdx.x * 4;
    for (int i = tid; i < 4 * ND; i += 256)
        es[i / ND][i % ND] = emb[(size_t)(row0 + i / ND) * ND + (i % ND)];
    __syncthreads();

    const int r = tid >> 6;   // 0..3 row within block
    const int d = tid & 63;   // output dim
    float a1 = b1s[d], a2 = b2s[d];
    #pragma unroll
    for (int k = 0; k < ND; ++k) {
        const float e = es[r][k];
        a1 = fmaf(e, W1s[d * ND + k], a1);
        a2 = fmaf(e, W2s[d * ND + k], a2);
    }
    const int row = row0 + r;
    v1h[(size_t)row * ND + d] = (_Float16)gelu_exact(a1);
    v2h[(size_t)row * ND + d] = (_Float16)gelu_exact(a2);
}

// ---------------------------------------------------------------------------
// Kernel 2: adj = relu(v1 v2^T - v2 v1^T), f32, via v_wmma_f32_16x16x32_f16.
// One wave computes one 16x16 output tile; K=64 split into two K=32 chunks.
// A-operand layout (16-bit A 16x32, ISA 7.12.2): lane m=L%16, half=L/16,
//   VGPR0-3 hold K = half*8 + 0..7, VGPR4-7 hold K = 16 + half*8 + 0..7.
// B-operand layout (16-bit B 32x16): lane n=L%16, half=L/16,
//   VGPR0-3 hold K = half*8 + 0..7, VGPR4-7 hold K = 16 + half*8 + 0..7.
// Since B = (v-tile)^T, each lane reads a row of the v-matrix -> plain b128s.
// ---------------------------------------------------------------------------
__global__ void adj_kernel(const _Float16* __restrict__ v1h,
                           const _Float16* __restrict__ v2h,
                           float* __restrict__ out) {
    const int lane = threadIdx.x & 31;
    const int wave = threadIdx.x >> 5;
    const int tile = blockIdx.x * 8 + wave;     // 512*512 tiles total
    const int ti = tile >> 9;                   // i-tile
    const int tj = tile & 511;                  // j-tile
    const int ib = ti * 16, jb = tj * 16;
    const int m = lane & 15;
    const int half = lane >> 4;

    const _Float16* rA1 = v1h + (size_t)(ib + m) * ND;  // A of product 1: v1 rows
    const _Float16* rA2 = v2h + (size_t)(ib + m) * ND;  // A of product 2: v2 rows
    const _Float16* rB1 = v2h + (size_t)(jb + m) * ND;  // B of product 1: v2^T cols
    const _Float16* rB2 = v1h + (size_t)(jb + m) * ND;  // B of product 2: v1^T cols

    union U { v16h v; v8h h[2]; };
    U a10, a11, a20, a21, b10, b11, b20, b21;

    // A chunks: k-chunk0 = K[0..31], chunk1 = K[32..63]
    a10.h[0] = *(const v8h*)(rA1 + half * 8);
    a10.h[1] = *(const v8h*)(rA1 + 16 + half * 8);
    a11.h[0] = *(const v8h*)(rA1 + 32 + half * 8);
    a11.h[1] = *(const v8h*)(rA1 + 48 + half * 8);
    a20.h[0] = *(const v8h*)(rA2 + half * 8);
    a20.h[1] = *(const v8h*)(rA2 + 16 + half * 8);
    a21.h[0] = *(const v8h*)(rA2 + 32 + half * 8);
    a21.h[1] = *(const v8h*)(rA2 + 48 + half * 8);
    // B chunks: same K order as A (half*8 + 0..7, 16 + half*8 + 0..7)
    b10.h[0] = *(const v8h*)(rB1 + half * 8);
    b10.h[1] = *(const v8h*)(rB1 + 16 + half * 8);
    b11.h[0] = *(const v8h*)(rB1 + 32 + half * 8);
    b11.h[1] = *(const v8h*)(rB1 + 48 + half * 8);
    b20.h[0] = *(const v8h*)(rB2 + half * 8);
    b20.h[1] = *(const v8h*)(rB2 + 16 + half * 8);
    b21.h[0] = *(const v8h*)(rB2 + 32 + half * 8);
    b21.h[1] = *(const v8h*)(rB2 + 48 + half * 8);

    v8f acc1 = {};
    v8f acc2 = {};
    acc1 = __builtin_amdgcn_wmma_f32_16x16x32_f16(false, a10.v, false, b10.v,
                                                  (short)0, acc1, false, false);
    acc1 = __builtin_amdgcn_wmma_f32_16x16x32_f16(false, a11.v, false, b11.v,
                                                  (short)0, acc1, false, false);
    acc2 = __builtin_amdgcn_wmma_f32_16x16x32_f16(false, a20.v, false, b20.v,
                                                  (short)0, acc2, false, false);
    acc2 = __builtin_amdgcn_wmma_f32_16x16x32_f16(false, a21.v, false, b21.v,
                                                  (short)0, acc2, false, false);

    // C/D layout: VGPR r holds (M = r + 8*half, N = lane%16)
    #pragma unroll
    for (int r = 0; r < 8; ++r) {
        float x = acc1[r] - acc2[r];
        x = x > 0.0f ? x : 0.0f;
        out[(size_t)(ib + r + 8 * half) * NN + (jb + m)] = x;
    }
}

// ---------------------------------------------------------------------------
// Kernel 3: per-row top-32 of (adj + 0.01*noise), write adj*mask in place.
// One 256-thread block per row; thread t owns columns j = t + 256*k, k=0..31
// (coalesced). 32 rounds of exact argmax (tie-break: lowest index) via LDS.
// ---------------------------------------------------------------------------
__global__ void topk_kernel(const float* __restrict__ noise,
                            float* __restrict__ out) {
    __shared__ float rv[256];
    __shared__ int   rj[256];

    const int row = blockIdx.x;
    const int t = threadIdx.x;
    const float* arow = out   + (size_t)row * NN;
    const float* nrow = noise + (size_t)row * NN;

    float adjv[32], s[32];
    #pragma unroll
    for (int k = 0; k < 32; ++k) {
        const int j = t + 256 * k;
        adjv[k] = arow[j];
        s[k] = adjv[k] + 0.01f * nrow[j];
    }

    unsigned keep = 0u;   // also serves as "already selected" mask
    for (int it = 0; it < KSEL; ++it) {
        float bv = -INFINITY;
        int bj = NN;
        #pragma unroll
        for (int k = 0; k < 32; ++k) {
            if (!(keep & (1u << k))) {
                const int j = t + 256 * k;
                if (s[k] > bv || (s[k] == bv && j < bj)) { bv = s[k]; bj = j; }
            }
        }
        rv[t] = bv; rj[t] = bj;
        __syncthreads();
        for (int off = 128; off > 0; off >>= 1) {
            if (t < off) {
                const float ov = rv[t + off];
                const int   oj = rj[t + off];
                if (ov > rv[t] || (ov == rv[t] && oj < rj[t])) { rv[t] = ov; rj[t] = oj; }
            }
            __syncthreads();
        }
        const int wj = rj[0];
        __syncthreads();                 // rv/rj reused next round
        if ((wj & 255) == t) keep |= 1u << (wj >> 8);
    }

    #pragma unroll
    for (int k = 0; k < 32; ++k) {
        const int j = t + 256 * k;
        out[(size_t)row * NN + j] = ((keep >> k) & 1u) ? adjv[k] : 0.0f;
    }
}

// ---------------------------------------------------------------------------
extern "C" void kernel_launch(void* const* d_in, const int* in_sizes, int n_in,
                              void* d_out, int out_size, void* d_ws, size_t ws_size,
                              hipStream_t stream) {
    const float* node_emb = (const float*)d_in[0];
    const float* noise    = (const float*)d_in[1];
    const float* W1       = (const float*)d_in[2];
    const float* b1       = (const float*)d_in[3];
    const float* W2       = (const float*)d_in[4];
    const float* b2       = (const float*)d_in[5];
    float* out = (float*)d_out;

    _Float16* v1h = (_Float16*)d_ws;                      // 8192*64 f16 = 1 MB
    _Float16* v2h = v1h + (size_t)NN * ND;                // +1 MB

    // 1) MLP + GELU -> f16 activations (4 rows per block)
    mlp_kernel<<<NN / 4, 256, 0, stream>>>(node_emb, W1, b1, W2, b2, v1h, v2h);

    // 2) adj = relu(v1 v2^T - v2 v1^T): 512x512 tiles, 8 waves (tiles) / block
    adj_kernel<<<(512 * 512) / 8, 256, 0, stream>>>(v1h, v2h, out);

    // 3) per-row top-32 selection + masking (in place on d_out)
    topk_kernel<<<NN, 256, 0, stream>>>(noise, out);
}
